// DistillMoE_63264868270175
// MI455X (gfx1250) — matrix operand order repaired and checked by reading, not compiled, not run on hardware
//
#include <hip/hip_runtime.h>
#include <hip/hip_bf16.h>

// ---------------------------------------------------------------------------
// MoE with noisy top-2 routing for gfx1250 (MI455X).
//   out layout: [B*512] updates  ++  [B*8] gating   (f32)
//   d_ws      : We converted to bf16 and transposed to [e][n][k]  (4 MB)
// ---------------------------------------------------------------------------

typedef __attribute__((ext_vector_type(16))) __bf16 v16bf;
typedef __attribute__((ext_vector_type(2)))  __bf16 bf16x2;
typedef __attribute__((ext_vector_type(8)))  float  v8f;
typedef int v4i_g __attribute__((vector_size(16)));   // matches builtin param type

#define D_IN   512
#define D_OUT  512
#define NEXP   8

__device__ __forceinline__ unsigned int pack_bf16(float a, float b) {
  bf16x2 p;
  p.x = (__bf16)a;           // native v_cvt (RNE)
  p.y = (__bf16)b;
  return __builtin_bit_cast(unsigned int, p);
}

__device__ __forceinline__ unsigned int rotl32(unsigned int x, int r) {
  return (x << r) | (x >> (32 - r));
}

// JAX threefry2x32 (20 rounds), key = (0, 1234) for jax.random.key(1234)
__device__ __forceinline__ void threefry2x32(unsigned int k0, unsigned int k1,
                                             unsigned int x0, unsigned int x1,
                                             unsigned int& o0, unsigned int& o1) {
  const unsigned int ks2 = k0 ^ k1 ^ 0x1BD11BDAu;
  x0 += k0; x1 += k1;
#define TF_R(r) { x0 += x1; x1 = rotl32(x1, r); x1 ^= x0; }
  TF_R(13) TF_R(15) TF_R(26) TF_R(6)   x0 += k1;  x1 += ks2 + 1u;
  TF_R(17) TF_R(29) TF_R(16) TF_R(24)  x0 += ks2; x1 += k0  + 2u;
  TF_R(13) TF_R(15) TF_R(26) TF_R(6)   x0 += k0;  x1 += k1  + 3u;
  TF_R(17) TF_R(29) TF_R(16) TF_R(24)  x0 += k1;  x1 += ks2 + 4u;
  TF_R(13) TF_R(15) TF_R(26) TF_R(6)   x0 += ks2; x1 += k0  + 5u;
#undef TF_R
  o0 = x0; o1 = x1;
}

__device__ __forceinline__ float erfinv_f(float u) {
  float w = -__logf((1.0f - u) * (1.0f + u));
  float p;
  if (w < 5.0f) {
    w -= 2.5f;
    p = 2.81022636e-08f;
    p = fmaf(p, w, 3.43273939e-07f);
    p = fmaf(p, w, -3.52338770e-06f);
    p = fmaf(p, w, -4.39150654e-06f);
    p = fmaf(p, w, 2.18580870e-04f);
    p = fmaf(p, w, -1.25372503e-03f);
    p = fmaf(p, w, -4.17768164e-03f);
    p = fmaf(p, w, 2.46640727e-01f);
    p = fmaf(p, w, 1.50140941e+00f);
  } else {
    w = sqrtf(w) - 3.0f;
    p = -2.00214257e-04f;
    p = fmaf(p, w, 1.00950558e-04f);
    p = fmaf(p, w, 1.34934322e-03f);
    p = fmaf(p, w, -3.67342844e-03f);
    p = fmaf(p, w, 5.73950773e-03f);
    p = fmaf(p, w, -7.62246130e-03f);
    p = fmaf(p, w, 9.43887047e-03f);
    p = fmaf(p, w, 1.00167406e+00f);
    p = fmaf(p, w, 2.83297682e+00f);
  }
  return p * u;
}

// ---------------------------------------------------------------------------
// Prep: We [e][k][n] f32  ->  ws [e][n][k] bf16   (transpose + convert)
// ---------------------------------------------------------------------------
__global__ void __launch_bounds__(256)
prep_weights(const float* __restrict__ We, unsigned short* __restrict__ wt_g) {
  const unsigned int idx = blockIdx.x * 256 + threadIdx.x; // e:3 | k4:7 | n:9
  const int n  = idx & 511;
  const int k  = ((idx >> 9) & 127) * 4;
  const int e  = idx >> 16;
  const float* src = We + ((size_t)e << 18) + (size_t)k * D_OUT + n;
  uint2 v;
  v.x = pack_bf16(src[0],            src[D_OUT]);
  v.y = pack_bf16(src[2 * D_OUT],    src[3 * D_OUT]);
  *(uint2*)(wt_g + ((size_t)e << 18) + (size_t)n * D_IN + k) = v;
}

// ---------------------------------------------------------------------------
// Router: logits, noisy logits, top-2 softmax gates
// ---------------------------------------------------------------------------
__global__ void __launch_bounds__(256)
router_kernel(const float* __restrict__ x,
              const float* __restrict__ Wr, const float* __restrict__ br,
              const float* __restrict__ Wn, const float* __restrict__ bn,
              float* __restrict__ gating, int B) {
  __shared__ float sWr[D_IN * NEXP];
  __shared__ float sWn[D_IN * NEXP];
  const int tid = threadIdx.x;
  for (int i = tid; i < D_IN * NEXP; i += 256) {
    sWr[i] = Wr[i];
    sWn[i] = Wn[i];
  }
  __syncthreads();

  const int t = blockIdx.x * 256 + tid;
  const float* xr = x + (size_t)t * D_IN;

  float lr[NEXP], ln[NEXP];
#pragma unroll
  for (int e = 0; e < NEXP; ++e) { lr[e] = 0.0f; ln[e] = 0.0f; }

  for (int k = 0; k < D_IN; k += 4) {
    const float4 xv = *(const float4*)(xr + k);
#pragma unroll
    for (int e = 0; e < NEXP; ++e) {
      lr[e] = fmaf(xv.x, sWr[(k + 0) * NEXP + e], lr[e]);
      lr[e] = fmaf(xv.y, sWr[(k + 1) * NEXP + e], lr[e]);
      lr[e] = fmaf(xv.z, sWr[(k + 2) * NEXP + e], lr[e]);
      lr[e] = fmaf(xv.w, sWr[(k + 3) * NEXP + e], lr[e]);
      ln[e] = fmaf(xv.x, sWn[(k + 0) * NEXP + e], ln[e]);
      ln[e] = fmaf(xv.y, sWn[(k + 1) * NEXP + e], ln[e]);
      ln[e] = fmaf(xv.z, sWn[(k + 2) * NEXP + e], ln[e]);
      ln[e] = fmaf(xv.w, sWn[(k + 3) * NEXP + e], ln[e]);
    }
  }

  // noisy logits:  lr + normal * softplus(ln)
  const unsigned int N = (unsigned int)B * NEXP;
  const unsigned int H = N >> 1;
  float noisy[NEXP];
#pragma unroll
  for (int e = 0; e < NEXP; ++e) {
    const unsigned int i = (unsigned int)t * NEXP + e;
    const unsigned int j = (i < H) ? i : (i - H);
    unsigned int o0, o1;
    threefry2x32(0u, 1234u, j, j + H, o0, o1);
    const unsigned int bits = (i < H) ? o0 : o1;
    const unsigned int fb = (bits >> 9) | 0x3F800000u;      // [1,2)
    const float u01 = __builtin_bit_cast(float, fb) - 1.0f; // [0,1)
    const float lo = -0.99999994f;                          // nextafter(-1,0)
    float u = fmaf(u01, 1.0f - lo, lo);
    u = fmaxf(u, lo);
    const float nrm = 1.41421356237f * erfinv_f(u);
    const float z = lr[e] + br[e];
    const float zn = ln[e] + bn[e];
    const float sp = (zn > 20.0f) ? zn : log1pf(__expf(zn));
    noisy[e] = z + nrm * sp;
  }

  // top-2 (first index wins ties, as jax.lax.top_k)
  int i1 = 0;
  float v1 = noisy[0];
#pragma unroll
  for (int e = 1; e < NEXP; ++e)
    if (noisy[e] > v1) { v1 = noisy[e]; i1 = e; }
  int i2 = -1;
  float v2 = -3.4e38f;
#pragma unroll
  for (int e = 0; e < NEXP; ++e)
    if (e != i1 && noisy[e] > v2) { v2 = noisy[e]; i2 = e; }

  const float g1 = 1.0f / (1.0f + __expf(v2 - v1));
  const float g2 = 1.0f - g1;
#pragma unroll
  for (int e = 0; e < NEXP; ++e) {
    float g = 0.0f;
    if (e == i1) g = g1;
    else if (e == i2) g = g2;
    gating[(size_t)t * NEXP + e] = g;
  }
}

// ---------------------------------------------------------------------------
// Expert GEMMs: out[m, n] = sum_e g[m,e] * (x[m,:] @ We[e][:,n] + be[e][n])
// bf16 WMMA 16x16x32, f32 accumulate.  Block tile 128(M) x 128(N), K=512.
// x strip (128x512 bf16, ~130 KB) staged once in LDS, reused for all 8 experts.
// W tiles: async global->LDS DMA, double-buffered one tile ahead.
// ---------------------------------------------------------------------------
#define XS_STRIDE 520   // bf16 elements per x row in LDS (16B-aligned, staggered)
#define WT_STRIDE 40    // bf16 elements per n-column of the 32-deep W tile
#define WT_ELEMS  (128 * WT_STRIDE)

__device__ __forceinline__ void async_copy_b128(const unsigned short* src,
                                                unsigned short* dst) {
#if __has_builtin(__builtin_amdgcn_global_load_async_to_lds_b128)
  __builtin_amdgcn_global_load_async_to_lds_b128(
      (__attribute__((address_space(1))) v4i_g*)(unsigned short*)src,
      (__attribute__((address_space(3))) v4i_g*)dst, 0, 0);
#else
  *(uint4*)dst = *(const uint4*)src;
#endif
}

__device__ __forceinline__ void wait_async() {
#if __has_builtin(__builtin_amdgcn_s_wait_asynccnt)
  __builtin_amdgcn_s_wait_asynccnt(0);
#else
  asm volatile("s_wait_asynccnt 0" ::: "memory");
#endif
}

__global__ void __launch_bounds__(256)
moe_kernel(const float* __restrict__ x, const unsigned short* __restrict__ Wt_g,
           const float* __restrict__ be, const float* __restrict__ gating,
           float* __restrict__ out, int B) {
  extern __shared__ char smem[];
  unsigned short* xs  = (unsigned short*)smem;                          // [128][XS_STRIDE]
  unsigned short* wtb = (unsigned short*)(smem + 128 * XS_STRIDE * 2);  // 2x [128][WT_STRIDE]
  float* gl = (float*)(smem + 128 * XS_STRIDE * 2 + 2 * WT_ELEMS * 2);  // [128][8]

  const int tid  = threadIdx.x;
  const int lane = tid & 31;
  const int wave = tid >> 5;
  const int wm   = wave & 3;   // wave M offset = wm*32
  const int wn   = wave >> 2;  // wave N offset = wn*64
  const int ln16 = lane & 15;
  const int lhi  = lane >> 4;  // 0|1 selects K-half per ISA fragment layout
  const int m0   = blockIdx.x * 128;
  const int n0   = blockIdx.y * 128;

  // per-thread source/dest for the per-tile W copy (2 x b128 each)
  const int cp_n   = tid >> 1;          // 0..127
  const int cp_seg = (tid & 1) * 2;     // segments {0,1} or {2,3} of 8 bf16
  const unsigned short* wsrc0 =
      Wt_g + (size_t)n0 * D_IN + (size_t)cp_n * D_IN + cp_seg * 8;
  unsigned short* wdst0 = wtb + cp_n * WT_STRIDE + cp_seg * 8;

  // issue tile 0 DMA first: it flies under the x staging below
  async_copy_b128(wsrc0,     wdst0);
  async_copy_b128(wsrc0 + 8, wdst0 + 8);

  // stage gates for this M strip
  for (int i = tid; i < 128 * NEXP; i += 256)
    gl[i] = gating[(size_t)(m0 + (i >> 3)) * NEXP + (i & 7)];

  // stage x strip: 128 rows x 512 cols, f32 -> bf16 (native cvt)
  for (int i = tid; i < 128 * 128; i += 256) {
    const int r = i >> 7, c4 = (i & 127) * 4;
    const float4 v = *(const float4*)(x + (size_t)(m0 + r) * D_IN + c4);
    uint2 pk;
    pk.x = pack_bf16(v.x, v.y);
    pk.y = pack_bf16(v.z, v.w);
    *(uint2*)(xs + r * XS_STRIDE + c4) = pk;
  }

  v8f f[2][4];
#pragma unroll
  for (int tm = 0; tm < 2; ++tm)
#pragma unroll
    for (int tn = 0; tn < 4; ++tn)
#pragma unroll
      for (int j = 0; j < 8; ++j) f[tm][tn][j] = 0.0f;

  v8f c[2][4];

  // 128 tiles total: tile t -> expert e = t>>4, k0 = (t&15)*32
  for (int t = 0; t < NEXP * 16; ++t) {
    const int e  = t >> 4;
    const int k0 = (t & 15) << 5;

    if ((t & 15) == 0) {
#pragma unroll
      for (int tm = 0; tm < 2; ++tm)
#pragma unroll
        for (int tn = 0; tn < 4; ++tn)
#pragma unroll
          for (int j = 0; j < 8; ++j) c[tm][tn][j] = 0.0f;
    }

    wait_async();      // tile t resident in buffer t&1 (this wave's writes)
    __syncthreads();   // all waves' writes visible; prev reads of buf (t+1)&1 done

    // issue tile t+1 into the other buffer; overlaps with compute below
    if (t + 1 < NEXP * 16) {
      const int e1  = (t + 1) >> 4;
      const int k1  = ((t + 1) & 15) << 5;
      const unsigned short* src = wsrc0 + ((size_t)e1 << 18) + k1;
      unsigned short* dst = wdst0 + ((t + 1) & 1) * WT_ELEMS;
      async_copy_b128(src,     dst);
      async_copy_b128(src + 8, dst + 8);
    }

    const unsigned short* wbuf = wtb + (t & 1) * WT_ELEMS;

    // A fragments (16-bit A 16x32 layout, ISA 7.12.2)
    v16bf a[2], b[4];
#pragma unroll
    for (int tm = 0; tm < 2; ++tm) {
      const unsigned short* p = xs + (wm * 32 + tm * 16 + ln16) * XS_STRIDE + k0 + lhi * 8;
      union { uint4 q[2]; v16bf v; } tload;
      tload.q[0] = *(const uint4*)p;
      tload.q[1] = *(const uint4*)(p + 16);
      a[tm] = tload.v;
    }
    // B fragments (16-bit B 32x16 layout: col = lane%16, K = 8*(lane/16)+{0..7} and 16+8*(lane/16)+{0..7})
#pragma unroll
    for (int tn = 0; tn < 4; ++tn) {
      const unsigned short* p = wbuf + (wn * 64 + tn * 16 + ln16) * WT_STRIDE + lhi * 8;
      union { uint4 q[2]; v16bf v; } tload;
      tload.q[0] = *(const uint4*)p;
      tload.q[1] = *(const uint4*)(p + 16);
      b[tn] = tload.v;
    }
#pragma unroll
    for (int tm = 0; tm < 2; ++tm)
#pragma unroll
      for (int tn = 0; tn < 4; ++tn)
        c[tm][tn] = __builtin_amdgcn_wmma_f32_16x16x32_bf16(
            false, a[tm], false, b[tn], (short)0, c[tm][tn], false, false);

    if ((t & 15) == 15) {
      // fold bias + gate into running accumulator
      float bias[4];
#pragma unroll
      for (int tn = 0; tn < 4; ++tn)
        bias[tn] = be[e * D_OUT + n0 + wn * 64 + tn * 16 + ln16];
#pragma unroll
      for (int tm = 0; tm < 2; ++tm) {
        float gv[8];
#pragma unroll
        for (int j = 0; j < 8; ++j)
          gv[j] = gl[(wm * 32 + tm * 16 + j + lhi * 8) * NEXP + e];
#pragma unroll
        for (int tn = 0; tn < 4; ++tn)
#pragma unroll
          for (int j = 0; j < 8; ++j)
            f[tm][tn][j] = fmaf(gv[j], c[tm][tn][j] + bias[tn], f[tm][tn][j]);
      }
    }
  }

  // store (C/D layout: VGPR j -> row j (lanes 0-15) / j+8 (lanes 16-31), col = lane%16)
#pragma unroll
  for (int tm = 0; tm < 2; ++tm)
#pragma unroll
    for (int j = 0; j < 8; ++j) {
      const int row = m0 + wm * 32 + tm * 16 + j + lhi * 8;
      float* op = out + (size_t)row * D_OUT + n0 + wn * 64 + ln16;
#pragma unroll
      for (int tn = 0; tn < 4; ++tn) op[tn * 16] = f[tm][tn][j];
    }
}

// ---------------------------------------------------------------------------
extern "C" void kernel_launch(void* const* d_in, const int* in_sizes, int n_in,
                              void* d_out, int out_size, void* d_ws, size_t ws_size,
                              hipStream_t stream) {
  const float* x  = (const float*)d_in[0];
  const float* Wr = (const float*)d_in[1];
  const float* br = (const float*)d_in[2];
  const float* Wn = (const float*)d_in[3];
  const float* bn = (const float*)d_in[4];
  const float* We = (const float*)d_in[5];
  const float* be = (const float*)d_in[6];

  const int B = in_sizes[0] / D_IN;

  float* out    = (float*)d_out;
  float* gating = out + (size_t)B * D_OUT;
  unsigned short* Wt_g = (unsigned short*)d_ws;   // 4 MB bf16 [e][n][k]

  prep_weights<<<(NEXP * D_OUT * (D_IN / 4)) / 256, 256, 0, stream>>>(We, Wt_g);
  router_kernel<<<B / 256, 256, 0, stream>>>(x, Wr, br, Wn, bn, gating, B);

  const size_t smem = (size_t)(128 * XS_STRIDE + 2 * WT_ELEMS) * sizeof(unsigned short)
                    + (size_t)(128 * NEXP) * sizeof(float);
  dim3 grid(B / 128, D_OUT / 128);
  moe_kernel<<<grid, 256, smem, stream>>>(x, Wt_g, be, gating, out, B);
}
